// GCLayer_38268158607904
// MI455X (gfx1250) — matrix operand order repaired and checked by reading, not compiled, not run on hardware
//
#include <hip/hip_runtime.h>
#include <cstdint>
#include <cstddef>

// ---------------------------------------------------------------------------
// Types for CDNA5 WMMA (wave32): v16bf A/B fragments, v8f accumulator.
// ---------------------------------------------------------------------------
typedef __attribute__((ext_vector_type(16))) __bf16 v16bf;
typedef __attribute__((ext_vector_type(8)))  float  v8f;
typedef __attribute__((ext_vector_type(4)))  unsigned int tdm_v4u;
typedef __attribute__((ext_vector_type(8)))  int          tdm_v8i;
typedef __attribute__((ext_vector_type(4)))  int          tdm_v4i;

#if defined(__has_builtin)
#if __has_builtin(__builtin_amdgcn_tensor_load_to_lds)
#define HAVE_TDM 1
#endif
#endif

struct U16x16 { uint4 lo; uint4 hi; };   // 32 bytes == v16bf

__device__ __forceinline__ unsigned short f2bf(float f) {
  // round-to-nearest-even fp32 -> bf16
  unsigned u = __float_as_uint(f);
  u += 0x7FFFu + ((u >> 16) & 1u);
  return (unsigned short)(u >> 16);
}

__device__ __forceinline__ v16bf ld_frag(const unsigned short* p0,
                                         const unsigned short* p1) {
  U16x16 t;
  t.lo = *(const uint4*)p0;   // 16B LDS read (halves 0..7)
  t.hi = *(const uint4*)p1;   // 16B LDS read (halves 8..15)
  return __builtin_bit_cast(v16bf, t);
}

__device__ __forceinline__ void wait_tensorcnt0() {
#if defined(__has_builtin)
#if __has_builtin(__builtin_amdgcn_s_wait_tensorcnt)
  __builtin_amdgcn_s_wait_tensorcnt(0);
  return;
#else
  asm volatile("s_wait_tensorcnt 0x0" ::: "memory");
  return;
#endif
#else
  asm volatile("s_wait_tensorcnt 0x0" ::: "memory");
#endif
}

// ---------------------------------------------------------------------------
// K0: zero int buffer
// ---------------------------------------------------------------------------
__global__ void zero_i32(int* __restrict__ p, int n) {
  int i = blockIdx.x * blockDim.x + threadIdx.x;
  if (i < n) p[i] = 0;
}

// ---------------------------------------------------------------------------
// K1: degree histogram (int atomics -> deterministic)
// ---------------------------------------------------------------------------
__global__ void degree_kernel(const int* __restrict__ senders,
                              const int* __restrict__ receivers,
                              int* __restrict__ sdeg, int* __restrict__ rdeg,
                              int E) {
  int e = blockIdx.x * blockDim.x + threadIdx.x;
  if (e < E) {
    atomicAdd(&sdeg[senders[e]], 1);
    atomicAdd(&rdeg[receivers[e]], 1);
  }
}

// ---------------------------------------------------------------------------
// K1b: one-time W[k][n] fp32  ->  Wt[n][k] bf16 (K-contiguous rows).
//      Makes the GEMM B-tile a raw 2D copy -> TDM-eligible.
// ---------------------------------------------------------------------------
__global__ void convert_w_kernel(const float* __restrict__ W,
                                 unsigned short* __restrict__ Wt,
                                 int K, int Hdim) {
  int idx = blockIdx.x * blockDim.x + threadIdx.x;
  if (idx < K * Hdim) {
    const int n = idx / K;
    const int k = idx - n * K;
    Wt[(size_t)n * K + k] = f2bf(W[(size_t)k * Hdim + n]);
  }
}

// ---------------------------------------------------------------------------
// K2: H = (A @ W + b) * rsqrt(max(sdeg,1))   via v_wmma_f32_16x16x32_bf16
//   block = 256 threads = 8 wave32; tile = 32 rows x 256 cols
//   wave w: row-half rt=w&1 (16 rows), col-quad cq=w>>1 (4x 16-col tiles)
//   B tile (32k x 256n, bf16, stored [n][k]) is DMA'd by the Tensor Data
//   Mover from Wt each K-step (wave 0 issues, TENSORcnt-tracked).
// ---------------------------------------------------------------------------
#define TK 32

__global__ __launch_bounds__(256) void gcn_gemm_wmma(
    const float* __restrict__ A, const unsigned short* __restrict__ Wt,
    const float* __restrict__ bias, const int* __restrict__ sdeg,
    float* __restrict__ Hout, int N, int K, int Hdim) {
  __shared__ unsigned short As[32 * TK];    // [row][k]   bf16, 2 KB
  __shared__ unsigned short Bs[256 * TK];   // [n][k]     bf16, 16 KB

  const int tid  = threadIdx.x;
  const int wave = tid >> 5;
  const int lane = tid & 31;
  const int g    = lane >> 4;     // half-wave group (ISA layout tables)
  const int ln   = lane & 15;
  const int rt   = wave & 1;
  const int cq   = wave >> 1;
  const int rowbase = blockIdx.x * 32;

  v8f acc[4];
#pragma unroll
  for (int i = 0; i < 4; ++i) acc[i] = (v8f)(0.0f);

  const int a_row_ld = tid >> 3;          // 0..31
  const int a_k_ld   = (tid & 7) << 2;    // 0,4,...,28
  const int a_grow   = min(rowbase + a_row_ld, N - 1);
  const int arow     = 16 * rt + ln;

#if HAVE_TDM
  // TDM D# pieces that do not depend on k0 (all wave-uniform).
  const unsigned int lds_off = (unsigned int)(uintptr_t)(void*)&Bs[0];
  const unsigned int g1d0 = 1u << 16;                              // data_size=2B
  const unsigned int g1d1 = ((unsigned)K & 0xFFFFu) << 16;         // tensor_dim0 lo
  const unsigned int g1d2 = (((unsigned)K >> 16) & 0xFFFFu) |
                            (((unsigned)Hdim & 0xFFFFu) << 16);    // dim0 hi | dim1 lo
  const unsigned int g1d3 = (((unsigned)Hdim >> 16) & 0xFFFFu) |
                            ((unsigned)TK << 16);                  // dim1 hi | tile_dim0
  const unsigned int g1d4 = (unsigned)Hdim & 0xFFFFu;              // tile_dim1 (rows)
  const unsigned int g1d5 = (unsigned)K;                           // dim0_stride lo
  const tdm_v8i g1 = {(int)g1d0, (int)g1d1, (int)g1d2, (int)g1d3,
                      (int)g1d4, (int)g1d5, 0, 0};
  const tdm_v4i gz = {0, 0, 0, 0};
#if __clang_major__ >= 23
  const tdm_v8i gz8 = {0, 0, 0, 0, 0, 0, 0, 0};
#endif
#endif

  for (int k0 = 0; k0 < K; k0 += TK) {
    // cooperative A load: 32 rows x 32 k, one float4 per thread, coalesced
    {
      const float4 v = *(const float4*)(A + (size_t)a_grow * K + k0 + a_k_ld);
      unsigned short* d = &As[a_row_ld * TK + a_k_ld];
      d[0] = f2bf(v.x); d[1] = f2bf(v.y); d[2] = f2bf(v.z); d[3] = f2bf(v.w);
    }
#if HAVE_TDM
    // B tile via Tensor Data Mover: 2D tile (tile_dim0=32 k, tile_dim1=256 n)
    // from Wt (tensor_dim0=K, stride K), raw bf16 -> lands as Bs[n*32+k].
    if (wave == 0) {
      const uint64_t ga = (uint64_t)(uintptr_t)(const void*)Wt +
                          (uint64_t)k0 * 2u;                 // tile start
      const tdm_v4u g0 = {1u,                                // count=1
                          lds_off,                           // D#.lds_addr
                          (unsigned int)ga,                  // global_addr lo
                          ((unsigned int)((ga >> 32) & 0x01FFFFFFull)) |
                              0x80000000u};                  // addr hi | type=2
#if __clang_major__ >= 23
      __builtin_amdgcn_tensor_load_to_lds(g0, g1, gz, gz, gz8, 0);
#else
      __builtin_amdgcn_tensor_load_to_lds(g0, g1, gz, gz, 0);
#endif
      wait_tensorcnt0();   // TENSORcnt==0 before signaling the barrier
    }
#else
    // fallback: cooperative raw bf16 copy (thread t copies row n=t, 64 B)
    {
      const uint4* s = (const uint4*)(Wt + (size_t)tid * K + k0);
      uint4* d = (uint4*)&Bs[tid * TK];
      d[0] = s[0]; d[1] = s[1]; d[2] = s[2]; d[3] = s[3];
    }
#endif
    if (k0 + TK < K)
      __builtin_prefetch(A + (size_t)a_grow * K + k0 + TK + a_k_ld, 0, 0);
    __syncthreads();

    // A fragment: halves[0..7] <-> K = 8g.., halves[8..15] <-> K = 16+8g..
    v16bf afrag = ld_frag(&As[arow * TK + 8 * g],
                          &As[arow * TK + 16 + 8 * g]);
#pragma unroll
    for (int ct = 0; ct < 4; ++ct) {
      const int n = cq * 64 + ct * 16 + ln;
      // B fragment: halves[0..7] <-> K = 8g.., halves[8..15] <-> K = 16+8g.., N = ln
      v16bf bfrag = ld_frag(&Bs[n * TK + 8 * g],
                            &Bs[n * TK + 16 + 8 * g]);
      acc[ct] = __builtin_amdgcn_wmma_f32_16x16x32_bf16(
          false, afrag, false, bfrag, (short)0, acc[ct], false, false);
    }
    __syncthreads();
  }

  // epilogue: C/D layout -> row = r + 8g (+16*rt), col = ln (+tile base)
  float scale[8];
  int rows[8];
#pragma unroll
  for (int r = 0; r < 8; ++r) {
    rows[r] = rowbase + 16 * rt + 8 * g + r;
    const int d = (rows[r] < N) ? sdeg[rows[r]] : 1;
    scale[r] = rsqrtf(fmaxf((float)d, 1.0f));
  }
#pragma unroll
  for (int ct = 0; ct < 4; ++ct) {
    const int n = cq * 64 + ct * 16 + ln;
    const float bb = bias[n];
#pragma unroll
    for (int r = 0; r < 8; ++r) {
      if (rows[r] < N)
        Hout[(size_t)rows[r] * Hdim + n] = (acc[ct][r] + bb) * scale[r];
    }
  }
}

// ---------------------------------------------------------------------------
// K3: single-block exclusive scan of rdeg -> row_off[N+1], cursor[N]
// ---------------------------------------------------------------------------
__global__ __launch_bounds__(1024) void scan_kernel(
    const int* __restrict__ deg, int* __restrict__ row_off,
    int* __restrict__ cursor, int N) {
  __shared__ int sm[1024];
  __shared__ int running;
  const int t = threadIdx.x;
  if (t == 0) running = 0;
  __syncthreads();

  for (int base = 0; base < N; base += 1024) {
    const int i = base + t;
    const int v = (i < N) ? deg[i] : 0;
    const int runbase = running;
    sm[t] = v;
    __syncthreads();
#pragma unroll
    for (int off = 1; off < 1024; off <<= 1) {
      const int x = sm[t];
      const int y = (t >= off) ? sm[t - off] : 0;
      __syncthreads();
      sm[t] = x + y;
      __syncthreads();
    }
    const int incl = sm[t];
    const int blocksum = sm[1023];
    if (i < N) {
      const int o = runbase + incl - v;
      row_off[i] = o;
      cursor[i]  = o;
    }
    __syncthreads();
    if (t == 0) running = runbase + blocksum;
    __syncthreads();
  }
  if (t == 0) row_off[N] = running;
}

// ---------------------------------------------------------------------------
// K4: CSR bucket fill (3.2M int atomics total instead of 410M f32 atomics)
// ---------------------------------------------------------------------------
__global__ void csr_fill_kernel(const int* __restrict__ senders,
                                const int* __restrict__ receivers,
                                int* __restrict__ cursor,
                                int* __restrict__ csr_col, int E) {
  int e = blockIdx.x * blockDim.x + threadIdx.x;
  if (e < E) {
    const int r = receivers[e];
    const int pos = atomicAdd(&cursor[r], 1);
    csr_col[pos] = senders[e];
  }
}

// ---------------------------------------------------------------------------
// K5: per-node gather-sum. Bucket sorted by sender id -> bit-deterministic
//     fp32 accumulation order. All reads coalesced; h lives in L2 (102 MB).
// ---------------------------------------------------------------------------
#define MAXD 512

__global__ __launch_bounds__(256) void agg_kernel(
    const float* __restrict__ h, const int* __restrict__ row_off,
    const int* __restrict__ csr_col, float* __restrict__ out,
    int N, int Hdim) {
  __shared__ int list[MAXD];
  const int node = blockIdx.x;
  const int t = threadIdx.x;
  const int beg = row_off[node];
  const int end = row_off[node + 1];
  const int deg = end - beg;
  const int m = (deg < MAXD) ? deg : MAXD;

  for (int i = t; i < m; i += blockDim.x) list[i] = csr_col[beg + i];
  __syncthreads();
  if (t == 0) {               // small (avg ~16) insertion sort for determinism
    for (int i = 1; i < m; ++i) {
      const int key = list[i];
      int j = i - 1;
      while (j >= 0 && list[j] > key) { list[j + 1] = list[j]; --j; }
      list[j + 1] = key;
    }
  }
  __syncthreads();

  float acc = 0.0f;
  if (deg <= MAXD) {
    for (int i = 0; i < m; ++i)
      acc += h[(size_t)list[i] * Hdim + t];
  } else {                    // overflow fallback (unreachable at these stats)
    for (int i = beg; i < end; ++i)
      acc += h[(size_t)csr_col[i] * Hdim + t];
  }
  const float s = rsqrtf(fmaxf((float)deg, 1.0f));
  out[(size_t)node * Hdim + t] = acc * s;
}

// ---------------------------------------------------------------------------
// launch
// ---------------------------------------------------------------------------
extern "C" void kernel_launch(void* const* d_in, const int* in_sizes, int n_in,
                              void* d_out, int out_size, void* d_ws,
                              size_t ws_size, hipStream_t stream) {
  const int Hdim = in_sizes[2];               // 256
  const int K    = in_sizes[1] / Hdim;        // 512
  const int N    = in_sizes[0] / K;           // 100000
  const int E    = in_sizes[3];               // 1600000

  const float* nodes     = (const float*)d_in[0];
  const float* W         = (const float*)d_in[1];
  const float* bias      = (const float*)d_in[2];
  const int*   senders   = (const int*)d_in[3];
  const int*   receivers = (const int*)d_in[4];
  float*       out       = (float*)d_out;

  // workspace carve-up (~110.7 MB)
  char* ws = (char*)d_ws;
  float* h = (float*)ws;
  size_t off = ((size_t)N * Hdim * sizeof(float) + 255) & ~(size_t)255;
  unsigned short* Wt = (unsigned short*)(ws + off);
  off += ((size_t)K * Hdim * sizeof(unsigned short) + 255) & ~(size_t)255;
  int* sdeg    = (int*)(ws + off);
  int* rdeg    = sdeg + N;
  int* row_off = rdeg + N;          // N+1 entries
  int* cursor  = row_off + N + 1;
  int* csr_col = cursor + N;        // E entries

  // K0: zero both degree arrays (contiguous)
  {
    const int n = 2 * N;
    zero_i32<<<(n + 255) / 256, 256, 0, stream>>>(sdeg, n);
  }
  // K1: degrees
  degree_kernel<<<(E + 255) / 256, 256, 0, stream>>>(senders, receivers,
                                                     sdeg, rdeg, E);
  // K1b: one-time W -> bf16 transposed (TDM-friendly layout)
  {
    const int n = K * Hdim;
    convert_w_kernel<<<(n + 255) / 256, 256, 0, stream>>>(W, Wt, K, Hdim);
  }
  // K2: WMMA GEMM (B tile via Tensor Data Mover) fused with sender-side
  //     normalization + bias
  gcn_gemm_wmma<<<(N + 31) / 32, 256, 0, stream>>>(nodes, Wt, bias, sdeg, h,
                                                   N, K, Hdim);
  // K3: scan receiver degrees -> CSR offsets + fill cursors
  scan_kernel<<<1, 1024, 0, stream>>>(rdeg, row_off, cursor, N);
  // K4: CSR bucket fill
  csr_fill_kernel<<<(E + 255) / 256, 256, 0, stream>>>(senders, receivers,
                                                       cursor, csr_col, E);
  // K5: deterministic per-node gather + receiver-side normalization
  agg_kernel<<<N, Hdim, 0, stream>>>(h, row_off, csr_col, out, N, Hdim);
}
